// LSTMModel_42176578847246
// MI455X (gfx1250) — matrix operand order repaired and checked by reading, not compiled, not run on hardware
//
#include <hip/hip_runtime.h>

#define B_TOTAL 4096
#define T_STEPS 128
#define H_DIM   128
#define BM      16     // batch rows per block (one 16-row M tile)
#define NTILES  32     // 512 gate cols / 16

typedef __attribute__((ext_vector_type(16))) __bf16 v16bf;
typedef __attribute__((ext_vector_type(8)))  float  v8f;

__device__ __forceinline__ unsigned short f2bf(float f) {
  unsigned u = __builtin_bit_cast(unsigned, f);
  return (unsigned short)((u + 0x7FFFu + ((u >> 16) & 1u)) >> 16);
}
__device__ __forceinline__ float fast_rcp(float x) {
#if __has_builtin(__builtin_amdgcn_rcpf)
  return __builtin_amdgcn_rcpf(x);        // v_rcp_f32
#else
  return 1.0f / x;
#endif
}
__device__ __forceinline__ float fast_tanh(float x) {
#if __has_builtin(__builtin_amdgcn_tanhf)
  return __builtin_amdgcn_tanhf(x);       // gfx1250 v_tanh_f32
#elif __has_builtin(__builtin_amdgcn_tanh_f32)
  return __builtin_amdgcn_tanh_f32(x);
#else
  float e = __expf(2.0f * fminf(fmaxf(x, -10.0f), 10.0f));
  return 1.0f - 2.0f * fast_rcp(e + 1.0f);
#endif
}
__device__ __forceinline__ float sigmoidf(float x) {
  // sigmoid(x) = 0.5*tanh(x/2) + 0.5  -> one TRANS op + fma
  return fmaf(0.5f, fast_tanh(0.5f * x), 0.5f);
}

// Map K-index within a 32-wide chunk to the 16-bit A-fragment (khalf, slot).
// ISA 7.12.2: lanes 0-15 hold kk in {0..7,16..23}, lanes 16-31 hold {8..15,24..31}.
__device__ __forceinline__ void a_slot(int kk, int& khalf, int& s) {
  khalf = (kk >> 3) & 1;
  s = (kk & 7) + ((kk >> 4) & 1) * 8;
}

// One LSTM layer: block owns BM batch rows, loops all T timesteps internally.
// g = [x_t | h_{t-1}] @ [W_ih | W_hh]^T  via bf16 WMMA, f32 accum, cell in f32.
template <int KSTEPS, int KIN_PAD, int KIN_REAL, bool IN_F32>
__global__ __launch_bounds__(256) void lstm_scan(
    const float* __restrict__ xin_f32,          // layer0 input [B][T][KIN_REAL]
    const unsigned short* __restrict__ xin_bf,  // layer1/2 input [B][T][128] bf16
    const float* __restrict__ w_ih,             // [512][KIN_REAL]
    const float* __restrict__ w_hh,             // [512][128]
    const float* __restrict__ b_ih,             // [512]
    const float* __restrict__ b_hh,             // [512]
    unsigned short* __restrict__ h_out,         // [B][T][128] bf16 (or null)
    float* __restrict__ last_out)               // [B][128] f32 (or null)
{
  // Pre-swizzled WMMA fragments: 1 KB (512 u16) per 16x32 / 32x16 tile.
  __shared__ __align__(32) unsigned short Wfrag[NTILES * KSTEPS * 512];
  __shared__ __align__(32) unsigned short Afrag[KSTEPS * 512];

  const int tid  = threadIdx.x;
  const int lane = tid & 31;
  const int wv   = tid >> 5;               // wave 0..7: owns cols wv*16..wv*16+15 of each gate
  const int bbase = blockIdx.x * BM;

  // ---- one-time: swizzle combined [W_ih | W_hh] into B-fragment layout (bf16) ----
  // B frag (32x16): lane = khalf*16 + n ; slot s -> K = kstep*32 + khalf*8 + (s&7) + (s>>3)*16 ; B[k][n] = W[col][k]
  for (int idx = tid; idx < NTILES * KSTEPS * 32 * 16; idx += 256) {
    int s     = idx & 15;
    int ln    = (idx >> 4) & 31;
    int frag  = idx >> 9;
    int kstep = frag % KSTEPS;
    int ntile = frag / KSTEPS;
    int k     = kstep * 32 + (ln >> 4) * 8 + (s & 7) + (s >> 3) * 16;
    int col   = ntile * 16 + (ln & 15);
    float w;
    if (k < KIN_PAD) w = (k < KIN_REAL) ? w_ih[col * KIN_REAL + k] : 0.0f;
    else             w = w_hh[col * H_DIM + (k - KIN_PAD)];
    Wfrag[frag * 512 + ln * 16 + s] = f2bf(w);
  }
  // zero A fragments: h0 = 0, and K-padding stays 0 forever
  for (int idx = tid; idx < KSTEPS * 512; idx += 256) Afrag[idx] = 0;

  // per-lane bias for its gate column j (constant over t)
  const int j = wv * 16 + (lane & 15);
  float bias[4];
#pragma unroll
  for (int g = 0; g < 4; ++g) bias[g] = b_ih[g * H_DIM + j] + b_hh[g * H_DIM + j];

  float creg[8];
#pragma unroll
  for (int v = 0; v < 8; ++v) creg[v] = 0.0f;

  // ---- prologue: prefetch x_0 into registers ----
  const int rA  = tid >> 4;                 // bf16 path: row
  const int kbA = (tid & 15) * 8;           // bf16 path: 8 bf16 = one b128 load
  const int r0  = tid / KIN_REAL;           // f32 path (layer 0)
  const int k0  = tid % KIN_REAL;
  uint4 px = {0, 0, 0, 0};
  float pxf = 0.0f;
  if (IN_F32) {
    if (tid < BM * KIN_REAL)
      pxf = xin_f32[((size_t)(bbase + r0) * T_STEPS + 0) * KIN_REAL + k0];
  } else {
    px = *(const uint4*)(xin_bf + ((size_t)(bbase + rA) * T_STEPS + 0) * H_DIM + kbA);
  }

  __syncthreads();

  for (int t = 0; t < T_STEPS; ++t) {
    // ---- stage prefetched x_t into A fragments (K range [0, KIN_PAD)) ----
    if (IN_F32) {
      if (tid < BM * KIN_REAL) {
        int khalf, s; a_slot(k0 & 31, khalf, s);
        Afrag[(k0 >> 5) * 512 + (r0 + 16 * khalf) * 16 + s] = f2bf(pxf);
      }
    } else {
      unsigned e[4] = {px.x, px.y, px.z, px.w};
#pragma unroll
      for (int q = 0; q < 8; ++q) {
        unsigned short ev = (unsigned short)(e[q >> 1] >> ((q & 1) * 16));
        int k = kbA + q;
        int khalf, s; a_slot(k & 31, khalf, s);
        Afrag[(k >> 5) * 512 + (rA + 16 * khalf) * 16 + s] = ev;
      }
    }
    __syncthreads();

    // ---- prefetch x_{t+1} (global load overlaps the WMMA phase) ----
    {
      int tn = (t + 1 < T_STEPS) ? t + 1 : t;
      if (IN_F32) {
        if (tid < BM * KIN_REAL)
          pxf = xin_f32[((size_t)(bbase + r0) * T_STEPS + tn) * KIN_REAL + k0];
      } else {
        px = *(const uint4*)(xin_bf +
                             ((size_t)(bbase + rA) * T_STEPS + tn) * H_DIM + kbA);
      }
    }

    // ---- WMMA: gates = [x_t | h] @ Wcomb^T, wave wv covers n-tiles {g*8+wv} ----
    v8f acc[4] = {};
#pragma unroll
    for (int ks = 0; ks < KSTEPS; ++ks) {
      v16bf a = *(const v16bf*)&Afrag[ks * 512 + lane * 16];
#pragma unroll
      for (int g = 0; g < 4; ++g) {
        int ntile = g * 8 + wv;
        v16bf b = *(const v16bf*)&Wfrag[(ntile * KSTEPS + ks) * 512 + lane * 16];
        acc[g] = __builtin_amdgcn_wmma_f32_16x16x32_bf16(
            false, a, false, b, (short)0, acc[g], false, false);
      }
    }
    __syncthreads();   // all A-fragment reads done before h_t / x_{t+1} overwrite

    // ---- LSTM cell update: wave-local (i,f,g,o all in this wave's accumulators) ----
    // C/D layout: VGPR v, lane L -> row = v + 8*(L>>4), col = ntile*16 + (L&15)
#pragma unroll
    for (int v = 0; v < 8; ++v) {
      float gi = acc[0][v] + bias[0];
      float gf = acc[1][v] + bias[1];
      float gg = acc[2][v] + bias[2];
      float go = acc[3][v] + bias[3];
      float c  = sigmoidf(gf) * creg[v] + sigmoidf(gi) * fast_tanh(gg);
      float h  = sigmoidf(go) * fast_tanh(c);
      creg[v]  = c;

      int r = v + 8 * (lane >> 4);
      // write h_t into its A-fragment slot (K = KIN_PAD + j)
      int k = KIN_PAD + j;
      int khalf, s; a_slot(k & 31, khalf, s);
      unsigned short hb = f2bf(h);
      Afrag[(k >> 5) * 512 + (r + 16 * khalf) * 16 + s] = hb;
      if (h_out)
        h_out[((size_t)(bbase + r) * T_STEPS + t) * H_DIM + j] = hb;
      if (last_out && t == T_STEPS - 1)
        last_out[(size_t)(bbase + r) * H_DIM + j] = h;
    }
    // next iter: x_{t+1} staging (disjoint LDS region) then barrier -> WMMA sees h_t
  }
}

// out[b] = fc2_w . relu(fc1_w @ last[b] + fc1_b) + fc2_b
__global__ __launch_bounds__(64) void fc_head(
    const float* __restrict__ last, const float* __restrict__ fc1_w,
    const float* __restrict__ fc1_b, const float* __restrict__ fc2_w,
    const float* __restrict__ fc2_b, float* __restrict__ out)
{
  __shared__ float sl[128];
  __shared__ float part[64];
  int b = blockIdx.x, tid = threadIdx.x;
  sl[tid]      = last[(size_t)b * 128 + tid];
  sl[tid + 64] = last[(size_t)b * 128 + tid + 64];
  __syncthreads();
  float acc = fc1_b[tid];
#pragma unroll 4
  for (int k = 0; k < 128; ++k) acc += sl[k] * fc1_w[tid * 128 + k];
  part[tid] = fmaxf(acc, 0.0f) * fc2_w[tid];
  __syncthreads();
  for (int off = 32; off > 0; off >>= 1) {
    if (tid < off) part[tid] += part[tid + off];
    __syncthreads();
  }
  if (tid == 0) out[b] = part[0] + fc2_b[0];
}

extern "C" void kernel_launch(void* const* d_in, const int* in_sizes, int n_in,
                              void* d_out, int out_size, void* d_ws, size_t ws_size,
                              hipStream_t stream) {
  (void)in_sizes; (void)n_in; (void)out_size; (void)ws_size;
  const float* x     = (const float*)d_in[0];
  const float* w_ih0 = (const float*)d_in[1];
  const float* w_hh0 = (const float*)d_in[2];
  const float* b_ih0 = (const float*)d_in[3];
  const float* b_hh0 = (const float*)d_in[4];
  const float* w_ih1 = (const float*)d_in[5];
  const float* w_hh1 = (const float*)d_in[6];
  const float* b_ih1 = (const float*)d_in[7];
  const float* b_hh1 = (const float*)d_in[8];
  const float* w_ih2 = (const float*)d_in[9];
  const float* w_hh2 = (const float*)d_in[10];
  const float* b_ih2 = (const float*)d_in[11];
  const float* b_hh2 = (const float*)d_in[12];
  const float* fc1_w = (const float*)d_in[13];
  const float* fc1_b = (const float*)d_in[14];
  const float* fc2_w = (const float*)d_in[15];
  const float* fc2_b = (const float*)d_in[16];

  const size_t hbytes = (size_t)B_TOTAL * T_STEPS * H_DIM * sizeof(unsigned short);
  char* ws = (char*)d_ws;
  unsigned short* hA   = (unsigned short*)ws;
  unsigned short* hB   = (unsigned short*)(ws + hbytes);
  float*          last = (float*)(ws + 2 * hbytes);

  dim3 grid(B_TOTAL / BM), blk(256);
  // layer 0: K_in = 8 (padded to 32) -> KC = 160, 5 K-steps
  lstm_scan<5, 32, 8, true><<<grid, blk, 0, stream>>>(
      x, nullptr, w_ih0, w_hh0, b_ih0, b_hh0, hA, nullptr);
  // layer 1: KC = 256, 8 K-steps
  lstm_scan<8, 128, 128, false><<<grid, blk, 0, stream>>>(
      nullptr, hA, w_ih1, w_hh1, b_ih1, b_hh1, hB, nullptr);
  // layer 2: only final h needed
  lstm_scan<8, 128, 128, false><<<grid, blk, 0, stream>>>(
      nullptr, hB, w_ih2, w_hh2, b_ih2, b_hh2, nullptr, last);

  fc_head<<<dim3(B_TOTAL), dim3(64), 0, stream>>>(last, fc1_w, fc1_b, fc2_w, fc2_b,
                                                  (float*)d_out);
}
